// AttentionDecoder_33208687133060
// MI455X (gfx1250) — matrix operand order repaired and checked by reading, not compiled, not run on hardware
//
#include <hip/hip_runtime.h>
#include <hip/hip_bf16.h>

// ---------------------------------------------------------------------------
// Attention LSTM decoder (teacher-forced), CDNA5 / gfx1250.
// All GEMMs run on v_wmma_f32_16x16x32_bf16 with f32 accumulation; weights
// are converted to bf16 once per launch (L2-resident afterwards).
// Each GEMM wave owns one 16-col W tile and BOTH M tiles (M=32), so every
// W element is loaded from L2 exactly once per step.
// ---------------------------------------------------------------------------

typedef __attribute__((ext_vector_type(16))) __bf16 v16bf;
typedef __attribute__((ext_vector_type(8)))  float  v8f;
typedef __attribute__((ext_vector_type(4)))  unsigned int v4u;

#define B_   32
#define T_   128
#define LCTX 512
#define H_   512
#define E_   256
#define V_   8000
#define KG   1280   /* E + H + H : concatenated gate-GEMM K */
#define NG   2048   /* 4*H       : gate-GEMM N              */

__device__ __forceinline__ unsigned short f2bf(float f) {
  unsigned int x = __float_as_uint(f);
  x += 0x7FFFu + ((x >> 16) & 1u);          // round-to-nearest-even
  return (unsigned short)(x >> 16);
}

// ---------------- one-time preprocessing ----------------------------------

__global__ void k_f32_to_bf16(const float* __restrict__ src,
                              unsigned short* __restrict__ dst, int n) {
  int i = blockIdx.x * 256 + threadIdx.x;
  if (i < n) dst[i] = f2bf(src[i]);
}

__global__ void k_build_Wg(const float* __restrict__ W_ih,
                           const float* __restrict__ W_hh,
                           unsigned short* __restrict__ Wg) {
  int i = blockIdx.x * 256 + threadIdx.x;
  if (i >= NG * KG) return;
  int n = i / KG, k = i % KG;
  float v = (k < E_ + H_) ? W_ih[(size_t)n * (E_ + H_) + k]
                          : W_hh[(size_t)n * H_ + (k - E_ - H_)];
  Wg[i] = f2bf(v);
}

__global__ void k_build_bg(const float* __restrict__ b_ih,
                           const float* __restrict__ b_hh,
                           float* __restrict__ bg) {
  int i = blockIdx.x * 256 + threadIdx.x;
  if (i < NG) bg[i] = b_ih[i] + b_hh[i];
}

__global__ void k_zero(float* __restrict__ p, int n) {
  int i = blockIdx.x * 256 + threadIdx.x;
  if (i < n) p[i] = 0.f;
}

// ---------------- per-step elementwise kernels ----------------------------

// Assemble bf16 A-operand [32,1280] = [ embed[token] | ctx | h ]
__global__ void k_build_Ag(const long long* __restrict__ seq, int t,
                           const unsigned short* __restrict__ emb_bf,
                           const float* __restrict__ ctx,
                           const float* __restrict__ h,
                           unsigned short* __restrict__ Ag) {
  int i = blockIdx.x * 256 + threadIdx.x;
  if (i >= B_ * KG) return;
  int b = i / KG, k = i % KG;
  unsigned short v;
  if (k < E_) {
    int tok = (int)seq[(size_t)b * T_ + t];
    v = emb_bf[(size_t)tok * E_ + k];
  } else if (k < E_ + H_) {
    v = f2bf(ctx[b * H_ + (k - E_)]);
  } else {
    v = f2bf(h[b * H_ + (k - E_ - H_)]);
  }
  Ag[i] = v;
}

// LSTM cell: gates[32,2048] -> c,h (f32) + h (bf16) + A_om cols [512,1024)
__global__ void k_lstm(const float* __restrict__ gates,
                       float* __restrict__ c, float* __restrict__ h,
                       unsigned short* __restrict__ h_bf,
                       unsigned short* __restrict__ A_om) {
  int i = blockIdx.x * 256 + threadIdx.x;
  if (i >= B_ * H_) return;
  int b = i / H_, j = i % H_;
  const float* g = gates + (size_t)b * NG;
  float ig = g[j], fg = g[H_ + j], gg = g[2 * H_ + j], og = g[3 * H_ + j];
  float si = 1.f / (1.f + __expf(-ig));
  float sf = 1.f / (1.f + __expf(-fg));
  float so = 1.f / (1.f + __expf(-og));
  float cn = sf * c[i] + si * tanhf(gg);
  float hn = so * tanhf(cn);
  c[i] = cn;
  h[i] = hn;
  unsigned short hb = f2bf(hn);
  h_bf[i] = hb;
  A_om[(size_t)b * (2 * H_) + H_ + j] = hb;
}

// ---------------- bf16 WMMA GEMM  C[32,N] = A[32,K] * W[N,K]^T ------------
// 256 threads = 8 waves; wave w -> ntile = blockIdx.x*8 + w, and owns BOTH
// M tiles (two f32 accumulators) so each W tile is loaded exactly once.
// Per-lane operand gathers follow the CDNA5 bf16 16x16x32 layouts:
//   A: lane(l&15)=row M, K halves {kb..kb+7, 16+kb..16+kb+7}, kb=(l>>4)*8
//   B: lane(l&15)=col N, K halves {kb..kb+7, 16+kb..16+kb+7}, kb=(l>>4)*8
template <bool BIAS, bool TANH_ACT, bool WF32, bool WBF16>
__global__ void k_gemm_wmma(const unsigned short* __restrict__ A,
                            const unsigned short* __restrict__ W,
                            const float* __restrict__ bias,
                            float* __restrict__ Cf,
                            unsigned short* __restrict__ Cb,
                            int N, int K) {
  int lane  = threadIdx.x & 31;
  int wave  = threadIdx.x >> 5;
  int ntile = blockIdx.x * 8 + wave;
  if (ntile * 16 >= N) return;
  int row = lane & 15;
  int kbA = (lane >> 4) * 8;
  int kbB = (lane >> 4) * 8;
  const unsigned short* Arow0 = A + (size_t)row * K;               // M 0..15
  const unsigned short* Arow1 = A + (size_t)(16 + row) * K;        // M 16..31
  const unsigned short* Wrow  = W + (size_t)(ntile * 16 + row) * K;

  v8f acc0 = {};
  v8f acc1 = {};
  for (int k = 0; k < K; k += 32) {
    union { v16bf v; v4u q[2]; } a0, a1, b;
    b.q[0]  = *(const v4u*)(Wrow  + k + kbB);
    b.q[1]  = *(const v4u*)(Wrow  + k + kbB + 16);
    a0.q[0] = *(const v4u*)(Arow0 + k + kbA);
    a0.q[1] = *(const v4u*)(Arow0 + k + kbA + 16);
    a1.q[0] = *(const v4u*)(Arow1 + k + kbA);
    a1.q[1] = *(const v4u*)(Arow1 + k + kbA + 16);
    acc0 = __builtin_amdgcn_wmma_f32_16x16x32_bf16(
        false, a0.v, false, b.v, (short)0, acc0, false, false);
    acc1 = __builtin_amdgcn_wmma_f32_16x16x32_bf16(
        false, a1.v, false, b.v, (short)0, acc1, false, false);
  }

  int   n     = ntile * 16 + row;
  int   mbase = (lane >> 4) * 8;
  float bv    = BIAS ? bias[n] : 0.f;
#pragma unroll
  for (int r = 0; r < 8; ++r) {
    float v0 = acc0[r] + bv;
    float v1 = acc1[r] + bv;
    if (TANH_ACT) { v0 = tanhf(v0); v1 = tanhf(v1); }
    size_t i0 = (size_t)(mbase + r) * N + n;
    size_t i1 = (size_t)(16 + mbase + r) * N + n;
    if (WF32)  { Cf[i0] = v0;      Cf[i1] = v1; }
    if (WBF16) { Cb[i0] = f2bf(v0); Cb[i1] = f2bf(v1); }
  }
}

// ---------------- attention: scores -> softmax -> context -----------------
// One block per batch element; cnn_feats stays f32 (L2-resident).
__global__ void k_attn(const float* __restrict__ feats,
                       const float* __restrict__ mapped,
                       unsigned short* __restrict__ A_om) {
  __shared__ float sm_map[H_];
  __shared__ float sm_a[LCTX];
  __shared__ float red[256];
  int b = blockIdx.x, tid = threadIdx.x;
  sm_map[tid]       = mapped[b * H_ + tid];
  sm_map[tid + 256] = mapped[b * H_ + tid + 256];
  __syncthreads();

  int lane = tid & 31, wave = tid >> 5;
  const float* fb = feats + (size_t)b * LCTX * H_;
  // scores: wave-cooperative row dots (coalesced over H)
  for (int l = wave; l < LCTX; l += 8) {
    const float* rowp = fb + (size_t)l * H_;
    float s = 0.f;
    for (int hh = lane; hh < H_; hh += 32) s += rowp[hh] * sm_map[hh];
    for (int off = 16; off > 0; off >>= 1) s += __shfl_xor(s, off, 32);
    if (lane == 0) sm_a[l] = s;
  }
  __syncthreads();
  // softmax over L=512
  float m = fmaxf(sm_a[tid], sm_a[tid + 256]);
  red[tid] = m; __syncthreads();
  for (int off = 128; off > 0; off >>= 1) {
    if (tid < off) red[tid] = fmaxf(red[tid], red[tid + off]);
    __syncthreads();
  }
  float mx = red[0]; __syncthreads();
  float e0 = __expf(sm_a[tid] - mx), e1 = __expf(sm_a[tid + 256] - mx);
  red[tid] = e0 + e1; __syncthreads();
  for (int off = 128; off > 0; off >>= 1) {
    if (tid < off) red[tid] += red[tid + off];
    __syncthreads();
  }
  float inv = 1.f / red[0]; __syncthreads();
  sm_a[tid] = e0 * inv; sm_a[tid + 256] = e1 * inv;
  __syncthreads();
  // context: coalesced weighted column sums
  float a0 = 0.f, a1 = 0.f;
  for (int l = 0; l < LCTX; ++l) {
    float wl = sm_a[l];
    const float* rowp = fb + (size_t)l * H_;
    a0 += wl * rowp[tid];
    a1 += wl * rowp[tid + 256];
  }
  A_om[(size_t)b * (2 * H_) + tid]       = f2bf(a0);
  A_om[(size_t)b * (2 * H_) + tid + 256] = f2bf(a1);
}

// ---------------- log-softmax over V=8000, write to output ----------------
__global__ void k_logsoftmax(const float* __restrict__ logits,
                             float* __restrict__ out, int t) {
  __shared__ float red[256];
  int b = blockIdx.x, tid = threadIdx.x;
  const float* row = logits + (size_t)b * V_;
  float m = -1e30f;
  for (int v = tid; v < V_; v += 256) m = fmaxf(m, row[v]);
  red[tid] = m; __syncthreads();
  for (int off = 128; off > 0; off >>= 1) {
    if (tid < off) red[tid] = fmaxf(red[tid], red[tid + off]);
    __syncthreads();
  }
  float mx = red[0]; __syncthreads();
  float s = 0.f;
  for (int v = tid; v < V_; v += 256) s += __expf(row[v] - mx);
  red[tid] = s; __syncthreads();
  for (int off = 128; off > 0; off >>= 1) {
    if (tid < off) red[tid] += red[tid + off];
    __syncthreads();
  }
  float lse = mx + __logf(red[0]);
  float* orow = out + ((size_t)b * (T_ - 1) + t) * V_;
  for (int v = tid; v < V_; v += 256) orow[v] = row[v] - lse;
}

// ---------------------------------------------------------------------------

extern "C" void kernel_launch(void* const* d_in, const int* in_sizes, int n_in,
                              void* d_out, int out_size, void* d_ws,
                              size_t ws_size, hipStream_t stream) {
  (void)in_sizes; (void)n_in; (void)out_size; (void)ws_size;
  const float*     cnn_feats = (const float*)d_in[0];
  const long long* seq       = (const long long*)d_in[1];   // int64 tokens
  const float*     embed     = (const float*)d_in[2];
  const float*     W_ih      = (const float*)d_in[3];
  const float*     b_ih      = (const float*)d_in[4];
  const float*     W_hh      = (const float*)d_in[5];
  const float*     b_hh      = (const float*)d_in[6];
  const float*     W_hm      = (const float*)d_in[7];
  const float*     W_om      = (const float*)d_in[8];
  const float*     W_logit   = (const float*)d_in[9];
  const float*     b_logit   = (const float*)d_in[10];
  float*           out       = (float*)d_out;

  unsigned char* w = (unsigned char*)d_ws;
  size_t off = 0;
  auto carve = [&](size_t bytes) {
    void* p = w + off;
    off += (bytes + 255) & ~(size_t)255;
    return p;
  };
  unsigned short* Wg    = (unsigned short*)carve((size_t)NG * KG * 2);
  unsigned short* Whm   = (unsigned short*)carve((size_t)H_ * H_ * 2);
  unsigned short* Wom   = (unsigned short*)carve((size_t)H_ * 2 * H_ * 2);
  unsigned short* Wlg   = (unsigned short*)carve((size_t)V_ * H_ * 2);
  unsigned short* Emb   = (unsigned short*)carve((size_t)V_ * E_ * 2);
  float*          bg    = (float*)carve((size_t)NG * 4);
  unsigned short* Ag    = (unsigned short*)carve((size_t)B_ * KG * 2);
  float*          gates = (float*)carve((size_t)B_ * NG * 4);
  float*          hbuf  = (float*)carve((size_t)B_ * H_ * 4);
  float*          cbuf  = (float*)carve((size_t)B_ * H_ * 4);
  float*          ctx   = (float*)carve((size_t)B_ * H_ * 4);
  unsigned short* hbf   = (unsigned short*)carve((size_t)B_ * H_ * 2);
  float*          mapd  = (float*)carve((size_t)B_ * H_ * 4);
  unsigned short* Aom   = (unsigned short*)carve((size_t)B_ * 2 * H_ * 2);
  unsigned short* outbf = (unsigned short*)carve((size_t)B_ * H_ * 2);
  float*          logit = (float*)carve((size_t)B_ * V_ * 4);

  // one-time weight prep (bf16) + zero recurrent state
  k_build_Wg<<<(NG * KG + 255) / 256, 256, 0, stream>>>(W_ih, W_hh, Wg);
  k_build_bg<<<(NG + 255) / 256, 256, 0, stream>>>(b_ih, b_hh, bg);
  k_f32_to_bf16<<<(H_ * H_ + 255) / 256, 256, 0, stream>>>(W_hm, Whm, H_ * H_);
  k_f32_to_bf16<<<(H_ * 2 * H_ + 255) / 256, 256, 0, stream>>>(W_om, Wom, H_ * 2 * H_);
  k_f32_to_bf16<<<(V_ * H_ + 255) / 256, 256, 0, stream>>>(W_logit, Wlg, V_ * H_);
  k_f32_to_bf16<<<(V_ * E_ + 255) / 256, 256, 0, stream>>>(embed, Emb, V_ * E_);
  k_zero<<<(B_ * H_ + 255) / 256, 256, 0, stream>>>(hbuf, B_ * H_);
  k_zero<<<(B_ * H_ + 255) / 256, 256, 0, stream>>>(cbuf, B_ * H_);
  k_zero<<<(B_ * H_ + 255) / 256, 256, 0, stream>>>(ctx, B_ * H_);

  // N/16 column tiles, 8 waves (tiles) per block
  const int gb_gates = (NG / 16 + 7) / 8;   // 16
  const int gb_h     = (H_ / 16 + 7) / 8;   // 4
  const int gb_v     = (V_ / 16 + 7) / 8;   // 63

  for (int t = 0; t < T_ - 1; ++t) {
    k_build_Ag<<<(B_ * KG + 255) / 256, 256, 0, stream>>>(seq, t, Emb, ctx, hbuf, Ag);
    // gates = Ag @ Wg^T + (b_ih + b_hh)
    k_gemm_wmma<true, false, true, false>
        <<<gb_gates, 256, 0, stream>>>(Ag, Wg, bg, gates, nullptr, NG, KG);
    k_lstm<<<(B_ * H_ + 255) / 256, 256, 0, stream>>>(gates, cbuf, hbuf, hbf, Aom);
    // mapped = h @ W_hm^T
    k_gemm_wmma<false, false, true, false>
        <<<gb_h, 256, 0, stream>>>(hbf, Whm, nullptr, mapd, nullptr, H_, H_);
    k_attn<<<B_, 256, 0, stream>>>(cnn_feats, mapd, Aom);
    // ctx(out) = tanh([cc,h] @ W_om^T)  (f32 carry + bf16 for logits)
    k_gemm_wmma<false, true, true, true>
        <<<gb_h, 256, 0, stream>>>(Aom, Wom, nullptr, ctx, outbf, H_, 2 * H_);
    // logits = out @ W_logit^T + b_logit
    k_gemm_wmma<true, false, true, false>
        <<<gb_v, 256, 0, stream>>>(outbf, Wlg, b_logit, logit, nullptr, V_, H_);
    k_logsoftmax<<<B_, 256, 0, stream>>>(logit, out, t);
  }
}
